// Net_86681029967930
// MI455X (gfx1250) — matrix operand order repaired and checked by reading, not compiled, not run on hardware
//
#include <hip/hip_runtime.h>
#include <hip/hip_bf16.h>

typedef __attribute__((ext_vector_type(16))) _Float16 v16h;
typedef __attribute__((ext_vector_type(8)))  float    v8f;

#define NI        128
#define NH        100
#define NO        2
#define STEPS     25
#define BETA      0.9f
#define THR       1.0f

#define WAVES           4
#define ROWS_PER_WAVE   16
#define ROWS_PER_BLOCK  (WAVES * ROWS_PER_WAVE)
#define BLOCK_THREADS   (WAVES * 32)

// Fragment-order sizes (halves): one fragment = 32 lanes * 16 halves = 512
#define FRAG_HALVES 512
#define ONE_F16     ((unsigned short)0x3C00)

// LDS layout (bytes). All operand data is pre-swizzled into fragment order so
// every per-lane operand slice is a contiguous 32B -> 2x ds_load_b128.
#define OFF_W2     0                      // f16 frag [7][4][32][16] = 28672
#define OFF_W3     28672                  // f16 frag [1][4][32][16] = 4096
#define OFF_UNION  32768                  // setup: W1 frag 28672 | loop: spike frags f16[WAVES][4][32][16] = 16384
#define OFF_CUR1   61440                  // f32 cfrag [WAVES][7][32][8] = 28672
#define LDS_BYTES  90112

// ---- WMMA helper ------------------------------------------------------------

__device__ __forceinline__ v8f wmma_f16(v16h a, v16h b, v8f c) {
  // D = A(16x32 f16) * B(32x16 f16) + C(16x16 f32)
  return __builtin_amdgcn_wmma_f32_16x16x32_f16(
      /*neg_a=*/false, a, /*neg_b=*/false, b,
      /*c_mod=*/(short)0, c, /*reuse_a=*/false, /*reuse_b=*/false);
}

// A-fragment element mapping (16x32 f16 per ISA):
//  lane L (L<16): row M=L,    element e -> K = e<8 ?  e   : 16+(e-8)
//  lane L (>=16): row M=L-16, element e -> K = e<8 ?  8+e : 24+(e-8)
// Inverse (producer scatters (row,K) into fragment order):
//  colm = K%32:  laneHi = (colm>>3)&1 ;  e = (colm&7) | ((colm>>1)&8)
// Note: K=112 -> kk=3, laneHi=0, e=8  (never touched by per-step writes, which
// only hit e<8 within kk=3) -> pinned to 1.0; biases live at W-row k=112.

// One-time: build A-fragments of x (fp32 global -> f16), scattered gather.
__device__ __forceinline__ v16h afrag_x_global(const float* xr /*16 rows, stride NI*/,
                                               int kOff) {
  const int lane = threadIdx.x & 31;
  const int m    = lane & 15;
  const int kb   = kOff + ((lane & 16) ? 8 : 0);
  const float* row = xr + m * NI;
  v16h a;
#pragma unroll
  for (int e = 0; e < 16; ++e) {
    const int k = (e < 8) ? (kb + e) : (kb + 16 + (e - 8));
    a[e] = (_Float16)row[k];
  }
  return a;
}

// ---- fused SNN kernel -------------------------------------------------------

__global__ __launch_bounds__(BLOCK_THREADS) void snn_lif3_kernel(
    const float* __restrict__ x,
    const float* __restrict__ W1, const float* __restrict__ b1,
    const float* __restrict__ W2, const float* __restrict__ b2,
    const float* __restrict__ W3, const float* __restrict__ b3,
    float* __restrict__ out, int bTot) {
  extern __shared__ char smem[];
  _Float16* sW2f  = (_Float16*)(smem + OFF_W2);     // [7][4][32][16]
  _Float16* sW3f  = (_Float16*)(smem + OFF_W3);     // [4][32][16]
  _Float16* sW1f  = (_Float16*)(smem + OFF_UNION);  // setup only
  _Float16* sSpkF = (_Float16*)(smem + OFF_UNION);  // loop only (reuse)
  unsigned short* sSpkU = (unsigned short*)(smem + OFF_UNION);
  float*    sCur  = (float*)(smem + OFF_CUR1);      // [WAVES][7][32][8]

  const int tid  = threadIdx.x;
  const int lane = tid & 31;
  const int wave = tid >> 5;
  const int col  = lane & 15;
  const int hi   = (lane >> 4) & 1;
  const int rowBase = blockIdx.x * ROWS_PER_BLOCK + wave * ROWS_PER_WAVE;

  // ---- cooperative weight staging, pre-swizzled into B-fragment order ----
  // flat half index i -> e = i&15, laneB = (i>>4)&31, kk = (i>>9)&3, t = i>>11
  // B layout: laneB holds column n = t*16 + (laneB&15),
//           K = kk*32 + ((laneB&16)?8:0) + e + (e&8)
  // Branch-free: clamp indices into range, select the value afterwards.
  for (int i = tid; i < 7 * 4 * FRAG_HALVES; i += BLOCK_THREADS) {
    const int e = i & 15, lb = (i >> 4) & 31, kk = (i >> 9) & 3, t = i >> 11;
    const int n  = t * 16 + (lb & 15);
    const int k  = kk * 32 + ((lb & 16) ? 8 : 0) + e + (e & 8);
    const int nc = (n < NH) ? n : 0;                 // clamped (safe) indices
    const int kc = (k < NH) ? k : 0;
    const float w1v = (n < NH) ? W1[nc * NI + k] : 0.0f;   // k always < NI
    const float w2w = W2[nc * NH + kc];
    const float b2w = b2[nc];
    float w2v = (n < NH && k < NH) ? w2w : 0.0f;
    w2v = (n < NH && k == 112) ? b2w : w2v;          // bias row
    sW1f[i] = (_Float16)w1v;
    sW2f[i] = (_Float16)w2v;
  }
  for (int i = tid; i < 4 * FRAG_HALVES; i += BLOCK_THREADS) {
    const int e = i & 15, lb = (i >> 4) & 31, kk = (i >> 9) & 3;
    const int n  = lb & 15;
    const int k  = kk * 32 + ((lb & 16) ? 8 : 0) + e + (e & 8);
    const int nc = (n < NO) ? n : 0;
    const int kc = (k < NH) ? k : 0;
    float w3v = (n < NO && k < NH) ? W3[nc * NH + kc] : 0.0f;
    w3v = (n < NO && k == 112) ? b3[nc] : w3v;
    sW3f[i] = (_Float16)w3v;
  }
  __syncthreads();

  // ---- cur1 = x @ W1^T + b1 (loop-invariant), stored in C-fragment order ----
  const float* xw = x + (size_t)rowBase * NI;
  v16h ax[4];
#pragma unroll
  for (int kk = 0; kk < 4; ++kk) ax[kk] = afrag_x_global(xw, kk * 32);

  float* myCur = sCur + wave * (7 * 32 * 8);
#pragma unroll
  for (int t = 0; t < 7; ++t) {
    const int n = t * 16 + col;
    const float bv = (n < NH) ? b1[n] : 0.0f;   // b1 folded here (one-time)
    v8f acc = wmma_f16(ax[0], *(const v16h*)(sW1f + ((t * 4 + 0) * 32 + lane) * 16),
                       (v8f){});
#pragma unroll
    for (int kk = 1; kk < 4; ++kk) {
      const v16h bf = *(const v16h*)(sW1f + ((t * 4 + kk) * 32 + lane) * 16);
      acc = wmma_f16(ax[kk], bf, acc);
    }
#pragma unroll
    for (int r = 0; r < 8; ++r) acc[r] += bv;
    *(v8f*)(myCur + (t * 32 + lane) * 8) = acc;   // contiguous 32B store
  }
  __syncthreads();                                // everyone done reading sW1f

  // init spike fragment scratch: zeros, except K==112 slots pinned to f16 1.0
  // (kk==3, laneA<16, e==8) which feed the bias rows of W2f/W3f.
  for (int i = tid; i < WAVES * 4 * FRAG_HALVES; i += BLOCK_THREADS) {
    const int e = i & 15, la = (i >> 4) & 31, kk = (i >> 9) & 3;
    sSpkU[i] = (kk == 3 && e == 8 && la < 16) ? ONE_F16 : (unsigned short)0;
  }
  __syncthreads();

  _Float16*       mySpkF = sSpkF + wave * (4 * FRAG_HALVES);
  unsigned short* mySpkU = sSpkU + wave * (4 * FRAG_HALVES);

  // Per-lane static scatter slots: spike (row = r+8*hi, colN = t*16+col) ->
  // half index ((kk*32 + laneA)*16 + e); +16 per extra row r.
  int spkSlot[7];
#pragma unroll
  for (int t = 0; t < 7; ++t) {
    const int colN = t * 16 + col;
    const int kk = colN >> 5, colm = colN & 31;
    const int laneA0 = 8 * hi + 16 * ((colm >> 3) & 1);
    const int e = (colm & 7) | ((colm >> 1) & 8);
    spkSlot[t] = (kk * 32 + laneA0) * 16 + e;
  }

  // ---- persistent LIF state in VGPRs ----
  v8f zero8;
#pragma unroll
  for (int r = 0; r < 8; ++r) zero8[r] = 0.0f;
  v8f m1[7], m2[7], m3 = zero8;
#pragma unroll
  for (int t = 0; t < 7; ++t) { m1[t] = zero8; m2[t] = zero8; }

  const size_t memOff = (size_t)STEPS * bTot * NO;

#pragma unroll 1
  for (int s = 0; s < STEPS; ++s) {
    // LIF1 (subtract reset from OLD mem, integrate, fire) -> s1 into A-frags
#pragma unroll
    for (int t = 0; t < 7; ++t) {
      const v8f cur = *(const v8f*)(myCur + (t * 32 + lane) * 8);
#pragma unroll
      for (int r = 0; r < 8; ++r) {
        float mem = m1[t][r];
        const float reset = (mem > THR) ? 1.0f : 0.0f;
        mem = BETA * mem + cur[r] - THR * reset;
        m1[t][r] = mem;
        mySpkU[spkSlot[t] + r * 16] = (mem > THR) ? ONE_F16 : (unsigned short)0;
      }
    }
    // contiguous A-fragment loads (same-wave DS ordering guarantees RAW)
    v16h as1[4];
#pragma unroll
    for (int kk = 0; kk < 4; ++kk)
      as1[kk] = *(const v16h*)(mySpkF + (kk * 32 + lane) * 16);

    // cur2 = s1 @ [W2^T ; b2] (bias via pinned K=112), C starts at inline 0
#pragma unroll
    for (int t = 0; t < 7; ++t) {
      v8f c2 = wmma_f16(as1[0], *(const v16h*)(sW2f + ((t * 4 + 0) * 32 + lane) * 16),
                        (v8f){});
#pragma unroll
      for (int kk = 1; kk < 4; ++kk) {
        const v16h bf = *(const v16h*)(sW2f + ((t * 4 + kk) * 32 + lane) * 16);
        c2 = wmma_f16(as1[kk], bf, c2);
      }
#pragma unroll
      for (int r = 0; r < 8; ++r) {
        float mem = m2[t][r];
        const float reset = (mem > THR) ? 1.0f : 0.0f;
        mem = BETA * mem + c2[r] - THR * reset;
        m2[t][r] = mem;
        mySpkU[spkSlot[t] + r * 16] = (mem > THR) ? ONE_F16 : (unsigned short)0;
      }
    }
    v16h as2[4];
#pragma unroll
    for (int kk = 0; kk < 4; ++kk)
      as2[kk] = *(const v16h*)(mySpkF + (kk * 32 + lane) * 16);

    // cur3 = s2 @ [W3^T ; b3] ; LIF3 ; record spk3/mem3
    v8f c3 = wmma_f16(as2[0], *(const v16h*)(sW3f + (0 * 32 + lane) * 16), (v8f){});
#pragma unroll
    for (int kk = 1; kk < 4; ++kk) {
      const v16h bf = *(const v16h*)(sW3f + (kk * 32 + lane) * 16);
      c3 = wmma_f16(as2[kk], bf, c3);
    }

    const size_t sbase = (size_t)s * bTot * NO;
#pragma unroll
    for (int r = 0; r < 8; ++r) {
      float mem = m3[r];
      const float reset = (mem > THR) ? 1.0f : 0.0f;
      mem = BETA * mem + c3[r] - THR * reset;
      m3[r] = mem;
      const float spk = (mem > THR) ? 1.0f : 0.0f;
      if (col < NO) {                       // lanes 0,1,16,17 hold the 2 outputs
        const size_t idx = sbase + (size_t)(rowBase + r + 8 * hi) * NO + col;
        out[idx]          = spk;
        out[memOff + idx] = mem;
      }
    }
  }
}

extern "C" void kernel_launch(void* const* d_in, const int* in_sizes, int n_in,
                              void* d_out, int out_size, void* d_ws, size_t ws_size,
                              hipStream_t stream) {
  (void)n_in; (void)out_size; (void)d_ws; (void)ws_size;
  const float* x  = (const float*)d_in[0];
  const float* W1 = (const float*)d_in[1];
  const float* b1 = (const float*)d_in[2];
  const float* W2 = (const float*)d_in[3];
  const float* b2 = (const float*)d_in[4];
  const float* W3 = (const float*)d_in[5];
  const float* b3 = (const float*)d_in[6];
  float* out = (float*)d_out;

  const int bTot   = in_sizes[0] / NI;              // 32768
  const int blocks = bTot / ROWS_PER_BLOCK;         // 512

  snn_lif3_kernel<<<blocks, BLOCK_THREADS, LDS_BYTES, stream>>>(
      x, W1, b1, W2, b2, W3, b3, out, bTot);
}
